// OFT_63797444214920
// MI455X (gfx1250) — matrix operand order repaired and checked by reading, not compiled, not run on hardware
//
#include <hip/hip_runtime.h>
#include <hip/hip_bf16.h>

// ---------------- problem constants (from reference) ----------------
#define B_   2
#define C_   256
#define H_   96
#define W_   320
#define L_   128
#define WG_  128
#define NL_  5              // int(160/32)
#define K_   (C_*NL_)       // 1280
#define LW_  (L_*WG_)       // 16384
#define M_   (B_*LW_)       // 32768
#define HW_  (H_*W_)        // 30720
#define NBOX (B_*NL_*LW_)   // 163840

typedef __attribute__((ext_vector_type(16))) __bf16 v16bf;
typedef __attribute__((ext_vector_type(8)))  float  v8f;

union FragBF { unsigned int u[8]; v16bf v; };

struct BoxTap {
    int   x0[4];
    int   y0[4];
    float wx1[4];
    float wy1[4];
    float scale;     // visible / area
    int   pad[3];    // -> 80 bytes
};

__device__ __forceinline__ unsigned short f32_to_bf16_rne(float f) {
    unsigned int u = __float_as_uint(f);
    u = (u + 0x7fffu + ((u >> 16) & 1u)) >> 16;
    return (unsigned short)u;
}

// ---------------- 1) x-cumsum + transpose to channel-last ----------------
// feat: (B,C,H,W) -> I: (B,H,W,C), I[b,y,x,c] = sum_{x'<=x} feat[b,c,y,x']
// block = 256 threads (one per channel), grid = B*H (one per row)
__global__ void oft_cumsum_x(const float* __restrict__ feat, float* __restrict__ I) {
    int b = blockIdx.x / H_;
    int y = blockIdx.x % H_;
    int c = threadIdx.x;
    const float* src = feat + (((size_t)b * C_ + c) * H_ + y) * W_;
    float*       dst = I + (((size_t)b * H_ + y) * W_) * C_ + c;
    float acc = 0.f;
    for (int x = 0; x < W_; ++x) {
        acc += src[x];
        dst[(size_t)x * C_] = acc;   // coalesced across c
    }
}

// ---------------- 2) y-cumsum in place ----------------
// grid = B*W (one per column), block = 256 (one per channel)
__global__ void oft_cumsum_y(float* __restrict__ I) {
    int b = blockIdx.x / W_;
    int x = blockIdx.x % W_;
    int c = threadIdx.x;
    float* p = I + (((size_t)b * H_) * W_ + x) * C_ + c;
    const size_t stride = (size_t)W_ * C_;
    float acc = 0.f;
    for (int y = 0; y < H_; ++y) {
        acc += p[y * stride];
        p[y * stride] = acc;
    }
}

// ---------------- 3) per-box projection + bilinear tap setup ----------------
__global__ void oft_box_setup(const float* __restrict__ calib,
                              const float* __restrict__ grid,
                              BoxTap* __restrict__ taps) {
    int gid = blockIdx.x * blockDim.x + threadIdx.x;
    if (gid >= NBOX) return;
    int b  = gid / (NL_ * LW_);
    int r  = gid % (NL_ * LW_);
    int nl = r / LW_;
    int lw = r % LW_;

    float gx = grid[lw * 3 + 0];
    float gy = grid[lw * 3 + 1];
    float gz = grid[lw * 3 + 2];
    float zoff = 32.f * (float)nl;
    const float* P = calib + b * 12;

    const float ox[8] = {-12.5f, 12.5f, 12.5f, -12.5f, -12.5f, 12.5f, 12.5f, -12.5f};
    const float oy[8] = {-12.5f, -12.5f, 12.5f, 12.5f, -12.5f, -12.5f, 12.5f, 12.5f};
    const float oz[8] = {0.f, 0.f, 0.f, 0.f, 32.f, 32.f, 32.f, 32.f};

    float xmin = 1e30f, ymin = 1e30f, xmax = -1e30f, ymax = -1e30f;
#pragma unroll
    for (int k = 0; k < 8; ++k) {
        float px = gx + ox[k], py = gy + oy[k], pz = gz + zoff + oz[k];
        float X = P[0]*px + P[1]*py + P[2] *pz + P[3];
        float Y = P[4]*px + P[5]*py + P[6] *pz + P[7];
        float Z = P[8]*px + P[9]*py + P[10]*pz + P[11];
        float Zc = fmaxf(Z, 1e-6f);
        float u = X / Zc, v = Y / Zc;
        float nx = fminf(fmaxf(2.f * u / (float)W_ - 1.f, -1.f), 0.95f);
        float ny = fminf(fmaxf(2.f * v / (float)H_ - 1.f, -1.f), 0.95f);
        xmin = fminf(xmin, nx); xmax = fmaxf(xmax, nx);
        ymin = fminf(ymin, ny); ymax = fmaxf(ymax, ny);
    }
    float area = (xmax - xmin) * (ymax - ymin) * (float)HW_ + 1e-6f;
    bool visible = (area > 1e-6f) && (area < (float)HW_ * 0.3f);

    BoxTap t;
    t.scale = visible ? (1.f / area) : 0.f;
    t.pad[0] = t.pad[1] = t.pad[2] = 0;
    // samples: lt(+), rb(+), rt(-), lb(-)
    float sxs[4] = {xmin, xmax, xmax, xmin};
    float sys[4] = {ymin, ymax, ymin, ymax};
#pragma unroll
    for (int s = 0; s < 4; ++s) {
        float X = ((sxs[s] + 1.f) * (float)W_ - 1.f) * 0.5f;
        float Y = ((sys[s] + 1.f) * (float)H_ - 1.f) * 0.5f;
        float x0f = floorf(X), y0f = floorf(Y);
        t.x0[s]  = (int)x0f;  t.y0[s]  = (int)y0f;
        t.wx1[s] = X - x0f;   t.wy1[s] = Y - y0f;
    }
    taps[gid] = t;
}

// ---------------- 4) Wc -> bf16 ----------------
__global__ void oft_cvt_wc(const float* __restrict__ Wc,
                           unsigned short* __restrict__ wcb, int n) {
    int i = blockIdx.x * blockDim.x + threadIdx.x;
    if (i < n) wcb[i] = f32_to_bf16_rne(Wc[i]);
}

// ---------------- 5) integral-image gather -> vox (bf16, M x K) ----------------
// block-per-box, thread-per-channel; each tap is a 1KB coalesced read.
// vox row m = b*LW + lw, col k = c*NL + nl
__global__ void oft_gather(const float* __restrict__ I,
                           const BoxTap* __restrict__ taps,
                           unsigned short* __restrict__ voxb) {
    int box = blockIdx.x;
    int c   = threadIdx.x;
    int b   = box / (NL_ * LW_);
    int r   = box % (NL_ * LW_);
    int nl  = r / LW_;
    int lw  = r % LW_;

    BoxTap t = taps[box];                       // uniform (scalar) loads
    const float* Ib = I + (size_t)b * H_ * W_ * C_ + c;

    float acc = 0.f;
#pragma unroll
    for (int s = 0; s < 4; ++s) {
        float sgn = (s < 2) ? 1.f : -1.f;
        int   x0  = t.x0[s], y0 = t.y0[s];
        float wx1 = t.wx1[s], wy1 = t.wy1[s];
        float wx0 = 1.f - wx1, wy0 = 1.f - wy1;
        int   xs[4] = {x0, x0 + 1, x0,     x0 + 1};
        int   ys[4] = {y0, y0,     y0 + 1, y0 + 1};
        float ws[4] = {wx0 * wy0, wx1 * wy0, wx0 * wy1, wx1 * wy1};
#pragma unroll
        for (int q = 0; q < 4; ++q) {
            int xx = xs[q], yy = ys[q];
            if (xx >= 0 && xx < W_ && yy >= 0 && yy < H_)
                acc += sgn * ws[q] * Ib[((size_t)yy * W_ + xx) * C_];
        }
    }
    acc *= t.scale;
    int m = b * LW_ + lw;
    voxb[(size_t)m * K_ + c * NL_ + nl] = f32_to_bf16_rne(acc);
}

// ---------------- 6) GEMM: out = relu(vox @ Wc^T + bc), via bf16 WMMA ----------------
// M=32768, N=256, K=1280. Each wave: one 16(M) x 128(N) strip (8 n-tiles).
// A frag (16x32 bf16, ISA 7.12.2): lane (half=l>>4, m=l&15), VGPR v holds
//   k = 16*(v>>2) + 8*half + 2*(v&3) + {0,1}   (packed pair -> one dword load)
// B frag (32x16 bf16): lane n=l&15, VGPR v holds
//   k = 16*(v>>2) + 8*half + 2*(v&3) + {0,1};  B[k,n] = Wc[n,k] (contiguous pair in Wc row)
__global__ void __launch_bounds__(256)
oft_gemm(const unsigned short* __restrict__ voxb,
         const unsigned short* __restrict__ wcb,
         const float* __restrict__ bc,
         float* __restrict__ out) {
    int wave = threadIdx.x >> 5;
    int lane = threadIdx.x & 31;
    int g    = blockIdx.x * 8 + wave;     // 0..4095
    int mt   = g >> 1;                    // m-tile 0..2047
    int nh   = g & 1;                     // n half: tiles [nh*8, nh*8+8)
    int half = lane >> 4;
    int l16  = lane & 15;

    v8f acc[8];
#pragma unroll
    for (int t = 0; t < 8; ++t)
#pragma unroll
        for (int e = 0; e < 8; ++e) acc[t][e] = 0.f;

    const unsigned short* arow = voxb + (size_t)(mt * 16 + l16) * K_;

#pragma unroll 1
    for (int ks = 0; ks < K_ / 32; ++ks) {
        int kbase = ks * 32;
        FragBF a;
#pragma unroll
        for (int v = 0; v < 8; ++v) {
            int k = kbase + 16 * (v >> 2) + 8 * half + 2 * (v & 3);
            a.u[v] = *(const unsigned int*)(arow + k);
        }
#pragma unroll
        for (int t = 0; t < 8; ++t) {
            int n = (nh * 8 + t) * 16 + l16;
            const unsigned short* brow = wcb + (size_t)n * K_;
            FragBF bf;
#pragma unroll
            for (int v = 0; v < 8; ++v) {
                int k = kbase + 16 * (v >> 2) + 8 * half + 2 * (v & 3);
                bf.u[v] = *(const unsigned int*)(brow + k);
            }
            acc[t] = __builtin_amdgcn_wmma_f32_16x16x32_bf16(
                false, a.v, false, bf.v, (short)0, acc[t], false, false);
        }
    }

    // epilogue: C/D layout -> element (m = r + 8*half, n = l16) per n-tile
    int b      = (mt * 16) / LW_;          // each m-tile lies within one batch
    int lwbase = (mt * 16) - b * LW_ + 8 * half;
#pragma unroll
    for (int t = 0; t < 8; ++t) {
        int n = (nh * 8 + t) * 16 + l16;
        float bias = bc[n];
        float* op = out + (size_t)b * C_ * LW_ + (size_t)n * LW_ + lwbase;
#pragma unroll
        for (int rr = 0; rr < 8; ++rr)
            op[rr] = fmaxf(acc[t][rr] + bias, 0.f);
    }
}

// ---------------- launch ----------------
extern "C" void kernel_launch(void* const* d_in, const int* in_sizes, int n_in,
                              void* d_out, int out_size, void* d_ws, size_t ws_size,
                              hipStream_t stream) {
    (void)in_sizes; (void)n_in; (void)out_size; (void)ws_size;
    const float* feat  = (const float*)d_in[0];   // (B,C,H,W)
    const float* calib = (const float*)d_in[1];   // (B,3,4)
    const float* grid  = (const float*)d_in[2];   // (L,Wg,3)
    const float* Wc    = (const float*)d_in[3];   // (C, K)
    const float* bc    = (const float*)d_in[4];   // (C,)
    float* out = (float*)d_out;                   // (B,C,L,Wg)

    char* ws = (char*)d_ws;
    float*          I     = (float*)(ws);                          // 62,914,560 B
    unsigned short* voxb  = (unsigned short*)(ws + 62914560);      // 83,886,080 B
    unsigned short* wcb   = (unsigned short*)(ws + 146800640);     //    655,360 B
    BoxTap*         taps  = (BoxTap*)(ws + 147456000);             // 13,107,200 B

    oft_cumsum_x <<<B_ * H_, 256, 0, stream>>>(feat, I);
    oft_cumsum_y <<<B_ * W_, 256, 0, stream>>>(I);
    oft_box_setup<<<(NBOX + 255) / 256, 256, 0, stream>>>(calib, grid, taps);
    oft_cvt_wc   <<<(C_ * K_ + 255) / 256, 256, 0, stream>>>(Wc, wcb, C_ * K_);
    oft_gather   <<<NBOX, 256, 0, stream>>>(I, taps, voxb);
    oft_gemm     <<<512, 256, 0, stream>>>(voxb, wcb, bc, out);
}
